// MoeGatherRsOp_79920751444178
// MI455X (gfx1250) — compile-verified, matrix operand order checked by reading, not run on hardware
//
#include <hip/hip_runtime.h>
#include <hip/hip_bf16.h>
#include <stdint.h>

// Problem constants (from reference)
constexpr int NTOK = 8192;
constexpr int TOPK = 2;
constexpr int E    = 8;
constexpr int Kdim = 512;
constexpr int Ndim = 1024;
constexpr int Mdim = NTOK * TOPK; // 16384

// Tiling
constexpr int BM  = 128;
constexpr int BN  = 128;
constexpr int KC  = 64;   // K chunk staged in LDS per buffer
constexpr int LDA = 80;   // LDS row stride (elements) for A  (80*2=160B: 16B aligned rows)
constexpr int LDB = 80;   // LDS row stride (elements) for B^T

typedef __attribute__((ext_vector_type(16))) __bf16 v16bf;
typedef __attribute__((ext_vector_type(8)))  __bf16 v8bf;
typedef __attribute__((ext_vector_type(2)))  __bf16 v2bf;
typedef __attribute__((ext_vector_type(8)))  float  v8f;

// Pack two f32 into two bf16 (lo -> [15:0], hi -> [31:16]) using the packed
// CVT instruction when available (1 VALU per 2 elements), else scalar converts.
__device__ __forceinline__ uint32_t bfpair(float lo, float hi) {
#if __has_builtin(__builtin_amdgcn_cvt_pk_bf16_f32)
    v2bf r = __builtin_amdgcn_cvt_pk_bf16_f32(lo, hi);
    return __builtin_bit_cast(uint32_t, r);
#else
    __bf16 a = (__bf16)lo, b = (__bf16)hi;
    return (uint32_t)__builtin_bit_cast(uint16_t, a) |
           ((uint32_t)__builtin_bit_cast(uint16_t, b) << 16);
#endif
}

// ---------------------------------------------------------------------------
// Kernel 1: per-row epilogue scale + inverse scatter index
// ---------------------------------------------------------------------------
__global__ void prep_kernel(const int* __restrict__ splits,
                            const int* __restrict__ scatter,
                            const float* __restrict__ in_scale,
                            const float* __restrict__ w_scale,
                            const float* __restrict__ vec_scale,
                            float* __restrict__ rowscale,
                            int* __restrict__ inv) {
    int i = blockIdx.x * blockDim.x + threadIdx.x;
    if (i >= Mdim) return;
    int acc = 0, e = 0;
#pragma unroll
    for (int j = 0; j < E; ++j) {
        acc += splits[j];
        if (i >= acc) e = j + 1;
    }
    if (e > E - 1) e = E - 1;
    rowscale[i] = in_scale[0] * w_scale[e] * vec_scale[i];
    inv[scatter[i]] = i / TOPK;   // scatter is a permutation of M
}

// ---------------------------------------------------------------------------
// Kernel 2: zero the output (atomic-reduce target; harness poisons d_out)
// ---------------------------------------------------------------------------
__global__ void zero_out_kernel(float4* __restrict__ out) {
    size_t i = (size_t)blockIdx.x * blockDim.x + threadIdx.x;
    out[i] = make_float4(0.f, 0.f, 0.f, 0.f);
}

// ---------------------------------------------------------------------------
// Kernel 3: grouped GEMM (bf16 WMMA, f32 accum) + fused scaled scatter-reduce
// ---------------------------------------------------------------------------
__global__ __launch_bounds__(256, 1)
void moe_gemm_kernel(const float* __restrict__ in,
                     const float* __restrict__ weight,
                     const int* __restrict__ splits,
                     const float* __restrict__ rowscale,
                     const int* __restrict__ inv,
                     float* __restrict__ out) {
    __shared__ __bf16 shA[2][BM * LDA];
    __shared__ __bf16 shB[2][BN * LDB];

    const int tile_n = blockIdx.x;
    const int tile_m = blockIdx.y;
    const int row0 = tile_m * BM;
    const int col0 = tile_n * BN;

    // expert owning this row tile (splits are multiples of BM here)
    int acc = 0, e = 0;
#pragma unroll
    for (int j = 0; j < E; ++j) {
        acc += splits[j];
        if (row0 >= acc) e = j + 1;
    }
    if (e > E - 1) e = E - 1;
    const float* __restrict__ Wp = weight + (size_t)e * Kdim * Ndim;

    const int tid  = threadIdx.x;
    const int lane = tid & 31;
    const int wave = tid >> 5;                // 8 waves
    const int wm   = wave & 3;                // 4 row-groups of 32
    const int wn   = wave >> 2;               // 2 col-groups of 64
    const int l15  = lane & 15;
    const int half = lane >> 4;               // 0 or 1
    const int chnk = half << 3;               // K base 0 or 8 (16-bit WMMA layout)

    v8f c[2][4] = {};                         // 2x4 grid of 16x16 f32 accumulators

    // --- stage A chunk: [BM x KC] f32 -> bf16 LDS (row-major, K contiguous)
    auto loadA = [&](int buf, int kk) {
        const int r    = tid >> 1;            // 0..127
        const int kseg = (tid & 1) << 5;      // 0 or 32
        const float4* src = reinterpret_cast<const float4*>(
            in + (size_t)(row0 + r) * Kdim + kk + kseg);
        __bf16* dst = &shA[buf][r * LDA + kseg];
#pragma unroll
        for (int u = 0; u < 8; ++u) {
            float4 v = src[u];
            uint2 w;
            w.x = bfpair(v.x, v.y);
            w.y = bfpair(v.z, v.w);
            *reinterpret_cast<uint2*>(dst + u * 4) = w;   // ds_store_b64
        }
    };

    // --- stage B chunk: [KC x BN] f32 -> bf16 LDS, TRANSPOSED ([n][k]).
    // Each thread owns a 4(k) x 8(n) micro-tile: coalesced b128 global loads
    // along n, aligned b64 LDS stores along k.
    auto loadB = [&](int buf, int kk) {
        const int k0 = (tid >> 4) << 2;       // 0,4,...,60
        const int n0 = (tid & 15) << 3;       // 0,8,...,120
        const float* src = Wp + (size_t)(kk + k0) * Ndim + col0 + n0;
        float rv[4][8];
#pragma unroll
        for (int kr = 0; kr < 4; ++kr) {
            const float4* p = reinterpret_cast<const float4*>(src + (size_t)kr * Ndim);
            float4 a = p[0], b = p[1];
            rv[kr][0] = a.x; rv[kr][1] = a.y; rv[kr][2] = a.z; rv[kr][3] = a.w;
            rv[kr][4] = b.x; rv[kr][5] = b.y; rv[kr][6] = b.z; rv[kr][7] = b.w;
        }
#pragma unroll
        for (int u = 0; u < 8; ++u) {
            uint2 w;
            w.x = bfpair(rv[0][u], rv[1][u]);   // k0, k0+1
            w.y = bfpair(rv[2][u], rv[3][u]);   // k0+2, k0+3
            *reinterpret_cast<uint2*>(&shB[buf][(n0 + u) * LDB + k0]) = w; // ds_store_b64
        }
    };

    // --- one K=32 WMMA step from LDS buffer `buf`, chunk offset `ks` (0 or 32)
    auto mma_step = [&](int buf, int ks) {
        v16bf a[2], b[4];
#pragma unroll
        for (int i = 0; i < 2; ++i) {
            const __bf16* p = &shA[buf][(wm * 32 + i * 16 + l15) * LDA + ks + chnk];
            v8bf lo = *(const v8bf*)p;          // K = chnk .. chnk+7
            v8bf hi = *(const v8bf*)(p + 16);   // K = chnk+16 .. chnk+23
#pragma unroll
            for (int q = 0; q < 8; ++q) { a[i][q] = lo[q]; a[i][q + 8] = hi[q]; }
        }
#pragma unroll
        for (int j = 0; j < 4; ++j) {
            const __bf16* p = &shB[buf][(wn * 64 + j * 16 + l15) * LDB + ks + chnk];
            v8bf lo = *(const v8bf*)p;
            v8bf hi = *(const v8bf*)(p + 16);
#pragma unroll
            for (int q = 0; q < 8; ++q) { b[j][q] = lo[q]; b[j][q + 8] = hi[q]; }
        }
#pragma unroll
        for (int i = 0; i < 2; ++i)
#pragma unroll
            for (int j = 0; j < 4; ++j)
                c[i][j] = __builtin_amdgcn_wmma_f32_16x16x32_bf16(
                    false, a[i], false, b[j], (short)0, c[i][j], false, false);
    };

    // --- main loop: double-buffered K streaming
    loadA(0, 0);
    loadB(0, 0);
    __syncthreads();
#pragma unroll 2
    for (int kk = 0; kk < Kdim; kk += KC) {
        const int buf = (kk / KC) & 1;
        if (kk + KC < Kdim) {
            loadA(buf ^ 1, kk + KC);
            loadB(buf ^ 1, kk + KC);
        }
        mma_step(buf, 0);
        mma_step(buf, 32);
        __syncthreads();
    }

    // --- epilogue: scale per row, fused top-k reduce via atomics on out[token]
    const int half8 = half * 8;
#pragma unroll
    for (int i = 0; i < 2; ++i) {
        const int rbase = row0 + wm * 32 + i * 16 + half8;
        float s[8];
        int   t[8];
#pragma unroll
        for (int v = 0; v < 8; ++v) {
            s[v] = rowscale[rbase + v];
            t[v] = inv[rbase + v];
        }
#pragma unroll
        for (int j = 0; j < 4; ++j) {
            const int cidx = col0 + wn * 64 + j * 16 + l15;
#pragma unroll
            for (int v = 0; v < 8; ++v)
                atomicAdd(&out[(size_t)t[v] * Ndim + cidx], c[i][j][v] * s[v]);
        }
    }
}

// ---------------------------------------------------------------------------
extern "C" void kernel_launch(void* const* d_in, const int* in_sizes, int n_in,
                              void* d_out, int out_size, void* d_ws, size_t ws_size,
                              hipStream_t stream) {
    (void)in_sizes; (void)n_in; (void)out_size; (void)ws_size;
    const float* inp       = (const float*)d_in[0];
    const float* weight    = (const float*)d_in[1];
    const int*   splits    = (const int*)d_in[2];
    const int*   scatter   = (const int*)d_in[3];
    const float* in_scale  = (const float*)d_in[4];
    const float* w_scale   = (const float*)d_in[5];
    const float* vec_scale = (const float*)d_in[6];
    float* out = (float*)d_out;

    float* rowscale = (float*)d_ws;                 // M floats
    int*   inv      = (int*)((char*)d_ws + (size_t)Mdim * sizeof(float)); // M ints

    prep_kernel<<<(Mdim + 255) / 256, 256, 0, stream>>>(
        splits, scatter, in_scale, w_scale, vec_scale, rowscale, inv);

    zero_out_kernel<<<(size_t)NTOK * Ndim / (256 * 4), 256, 0, stream>>>(
        (float4*)out);

    dim3 grid(Ndim / BN, Mdim / BM);                // (8, 128)
    moe_gemm_kernel<<<grid, 256, 0, stream>>>(inp, weight, splits,
                                              rowscale, inv, out);
}
